// MraSelfAttention_75496935129642
// MI455X (gfx1250) — matrix operand order repaired and checked by reading, not compiled, not run on hardware
//
#include <hip/hip_runtime.h>
#include <hip/hip_bf16.h>

// ---------------------------------------------------------------------------
// MRA attention for MI455X (gfx1250, wave32, WMMA).
// Pipeline: init -> QKV bf16-WMMA GEMM (register double-buffered) ->
//           block pooling -> low-res logits -> top-512 radix select ->
//           HR block phase0 (sparse max) -> HR block phase1 (exp + AV WMMA,
//           atomic scatter) -> final combine.
// ---------------------------------------------------------------------------

typedef __attribute__((ext_vector_type(16))) __bf16 v16bf;
typedef __attribute__((ext_vector_type(8)))  float  v8f;

#define MRA_B   4
#define MRA_S   4096
#define MRA_H   768
#define MRA_NH  12
#define MRA_BH  48          // B * NH
#define MRA_HD  64
#define MRA_NB  128         // S / 32 blocks
#define MRA_TOPK 512

__device__ __forceinline__ float bf2f(unsigned short u) {
    unsigned int x = ((unsigned int)u) << 16;
    return __builtin_bit_cast(float, x);
}
// Native convert: let the backend use the gfx1250 hardware f32->bf16 path
// instead of a bfe/add3 software RNE sequence.
__device__ __forceinline__ unsigned short f2bf(float f) {
    __bf16 h = (__bf16)f;
    return __builtin_bit_cast(unsigned short, h);
}
__device__ __forceinline__ __bf16 us2bf(unsigned short u) {
    return __builtin_bit_cast(__bf16, u);
}
__device__ __forceinline__ unsigned int fkey(float f) {
    unsigned int u = __builtin_bit_cast(unsigned int, f);
    return (u & 0x80000000u) ? ~u : (u | 0x80000000u);   // order-preserving
}
__device__ __forceinline__ void atomicMaxFloat(float* addr, float val) {
    if (val >= 0.f) atomicMax((int*)addr, __builtin_bit_cast(int, val));
    else            atomicMin((unsigned int*)addr, __builtin_bit_cast(unsigned int, val));
}
__device__ __forceinline__ void atomicMinFloat(float* addr, float val) {
    if (val >= 0.f) atomicMin((int*)addr, __builtin_bit_cast(int, val));
    else            atomicMax((unsigned int*)addr, __builtin_bit_cast(unsigned int, val));
}

// ---------------------------------------------------------------------------
// K0: initialize accumulators.
// ---------------------------------------------------------------------------
__global__ void mra_init(float* hr_out, float* hr_norm, float* max_vals, float* thresh) {
    size_t i = (size_t)blockIdx.x * blockDim.x + threadIdx.x;
    size_t stride = (size_t)gridDim.x * blockDim.x;
    const size_t nOut  = (size_t)MRA_BH * MRA_S * MRA_HD;
    const size_t nNorm = (size_t)MRA_BH * MRA_S;
    for (size_t j = i; j < nOut; j += stride) hr_out[j] = 0.f;
    for (size_t j = i; j < nNorm; j += stride) { hr_norm[j] = 0.f; max_vals[j] = -1e9f; }
    if (i < MRA_BH) thresh[i] = 3.0e38f;
}

// ---------------------------------------------------------------------------
// K1: QKV projection GEMM.  out[m, n] = X[m,:] . W[n%768, :] + bias, masked,
// written as bf16 in head layout (bh, s, d).  M=16384, N=2304, K=768.
// Workgroup: 256 threads = 8 waves, tile 128(M) x 64(N); wave tile 32x32
// (2x2 wmma 16x16x32 bf16 tiles), K step 32, global loads double-buffered
// through registers so they overlap the ds_load + v_wmma of the current tile.
// ---------------------------------------------------------------------------
__global__ __launch_bounds__(256) void mra_qkv_gemm(
    const float* __restrict__ X, const float* __restrict__ am,
    const float* __restrict__ Wq, const float* __restrict__ bq,
    const float* __restrict__ Wk, const float* __restrict__ bk,
    const float* __restrict__ Wv, const float* __restrict__ bv,
    unsigned short* __restrict__ qo, unsigned short* __restrict__ ko_,
    unsigned short* __restrict__ vo)
{
    __shared__ unsigned short As[128][40];   // [m][k] bf16, padded
    __shared__ unsigned short Bs[64][40];    // [n][k] bf16, padded

    const int tid = threadIdx.x;
    const int m0 = blockIdx.y * 128;
    const int n0 = blockIdx.x * 64;
    const int wid = tid >> 5, lane = tid & 31;
    const int lo = lane & 15, hi = lane >> 4;
    const int wm = wid & 3, wn = wid >> 2;

    // staging assignments
    const int ar = tid >> 1, ako = (tid & 1) * 16;       // A: 16 floats/thread
    const int bn = tid >> 2, bko = (tid & 3) * 8;        // B: 8 floats/thread
    const int gn = n0 + bn;
    const int bproj = gn / MRA_H, bnm = gn % MRA_H;
    const float* Wrow = ((bproj == 0) ? Wq : (bproj == 1) ? Wk : Wv) + (size_t)bnm * MRA_H;
    const float* Arow = X + (size_t)(m0 + ar) * MRA_H;

    float4 aReg[4], bReg[2];
#pragma unroll
    for (int j = 0; j < 4; ++j) aReg[j] = ((const float4*)(Arow + ako))[j];
#pragma unroll
    for (int j = 0; j < 2; ++j) bReg[j] = ((const float4*)(Wrow + bko))[j];

    v8f c[2][2] = {};

    for (int kt = 0; kt < MRA_H; kt += 32) {
        // convert staged registers to bf16 in LDS
#pragma unroll
        for (int j = 0; j < 4; ++j) {
            float4 f = aReg[j];
            As[ar][ako + 4 * j + 0] = f2bf(f.x);
            As[ar][ako + 4 * j + 1] = f2bf(f.y);
            As[ar][ako + 4 * j + 2] = f2bf(f.z);
            As[ar][ako + 4 * j + 3] = f2bf(f.w);
        }
#pragma unroll
        for (int j = 0; j < 2; ++j) {
            float4 f = bReg[j];
            Bs[bn][bko + 4 * j + 0] = f2bf(f.x);
            Bs[bn][bko + 4 * j + 1] = f2bf(f.y);
            Bs[bn][bko + 4 * j + 2] = f2bf(f.z);
            Bs[bn][bko + 4 * j + 3] = f2bf(f.w);
        }
        __syncthreads();

        // issue next tile's global loads; they overlap the wmma work below
        if (kt + 32 < MRA_H) {
#pragma unroll
            for (int j = 0; j < 4; ++j) aReg[j] = ((const float4*)(Arow + kt + 32 + ako))[j];
#pragma unroll
            for (int j = 0; j < 2; ++j) bReg[j] = ((const float4*)(Wrow + kt + 32 + bko))[j];
        }

        v16bf a[2], b[2];
#pragma unroll
        for (int mi = 0; mi < 2; ++mi) {
            int row = wm * 32 + mi * 16 + lo;
#pragma unroll
            for (int v = 0; v < 8; ++v) {
                int kk = 2 * v + ((v >= 4) ? 8 : 0) + hi * 8;   // A layout (16-bit 16x32)
                a[mi][2 * v]     = us2bf(As[row][kk]);
                a[mi][2 * v + 1] = us2bf(As[row][kk + 1]);
            }
        }
#pragma unroll
        for (int ni = 0; ni < 2; ++ni) {
            int col = wn * 32 + ni * 16 + lo;
#pragma unroll
            for (int v = 0; v < 8; ++v) {
                int kk = 2 * v + ((v >= 4) ? 8 : 0) + hi * 8;   // B layout (16-bit 32x16)
                b[ni][2 * v]     = us2bf(Bs[col][kk]);
                b[ni][2 * v + 1] = us2bf(Bs[col][kk + 1]);
            }
        }
#pragma unroll
        for (int mi = 0; mi < 2; ++mi)
#pragma unroll
            for (int ni = 0; ni < 2; ++ni)
                c[mi][ni] = __builtin_amdgcn_wmma_f32_16x16x32_bf16(
                    false, a[mi], false, b[ni], (short)0, c[mi][ni], false, false);
        __syncthreads();
    }

    // epilogue: bias + mask, scatter into per-head bf16 q/k/v
#pragma unroll
    for (int mi = 0; mi < 2; ++mi)
#pragma unroll
        for (int ni = 0; ni < 2; ++ni)
#pragma unroll
            for (int r = 0; r < 8; ++r) {
                int M = r + hi * 8, N = lo;
                int grow = m0 + wm * 32 + mi * 16 + M;
                int gcol = n0 + wn * 32 + ni * 16 + N;
                int bb = grow >> 12, s = grow & (MRA_S - 1);
                int proj = gcol / MRA_H, nm = gcol % MRA_H;
                int h = nm >> 6, d = nm & 63;
                const float* bias = (proj == 0) ? bq : (proj == 1) ? bk : bv;
                float mval = 1.f + am[(size_t)bb * MRA_S + s] * 1e-4f;
                float val = (c[mi][ni][r] + bias[nm]) * mval;
                unsigned short* dst = (proj == 0) ? qo : (proj == 1) ? ko_ : vo;
                dst[((size_t)(bb * MRA_NH + h) * MRA_S + s) * MRA_HD + d] = f2bf(val);
            }
}

// ---------------------------------------------------------------------------
// K2: block pooling -> tc, q_hat, k_hat, v_hat.
// grid: BH*128 blocks x 64 threads (thread = d).
// ---------------------------------------------------------------------------
__global__ __launch_bounds__(64) void mra_pool(
    const unsigned short* __restrict__ q, const unsigned short* __restrict__ k,
    const unsigned short* __restrict__ v, const float* __restrict__ am,
    float* __restrict__ tc, float* __restrict__ qh, float* __restrict__ kh,
    float* __restrict__ vh)
{
    int blk = blockIdx.x & (MRA_NB - 1);
    int bh  = blockIdx.x >> 7;
    int b   = bh / MRA_NH;
    int d   = threadIdx.x;
    float tcv = 0.f, sq = 0.f, sk = 0.f, sv = 0.f;
    for (int j = 0; j < 32; ++j) {
        int s = blk * 32 + j;
        tcv += 1.f + am[(size_t)b * MRA_S + s] * 1e-4f;
        size_t base = ((size_t)bh * MRA_S + s) * MRA_HD + d;
        sq += bf2f(q[base]); sk += bf2f(k[base]); sv += bf2f(v[base]);
    }
    float denom = tcv + 1e-6f;
    size_t o = ((size_t)bh * MRA_NB + blk) * MRA_HD + d;
    qh[o] = sq / denom; kh[o] = sk / denom; vh[o] = sv / denom;
    if (d == 0) tc[(size_t)bh * MRA_NB + blk] = tcv;
}

// ---------------------------------------------------------------------------
// K3: low-res logits lrl = q_hat @ k_hat^T / 8, row_max (pre-penalty),
// then mask penalty.  One block per bh, 256 threads (2 per row).
// ---------------------------------------------------------------------------
__global__ __launch_bounds__(256) void mra_lowres(
    const float* __restrict__ qh, const float* __restrict__ kh,
    const float* __restrict__ tc, float* __restrict__ lrl,
    float* __restrict__ rowmax)
{
    __shared__ float Qh[MRA_NB * MRA_HD];   // 32 KB
    __shared__ float rmx[256];
    int bh = blockIdx.x, t = threadIdx.x;
    for (int i = t; i < MRA_NB * MRA_HD; i += 256)
        Qh[i] = qh[(size_t)bh * MRA_NB * MRA_HD + i];
    __syncthreads();

    int row = t >> 1, coff = (t & 1) * 64;
    float* dst = lrl + ((size_t)bh * MRA_NB + row) * MRA_NB + coff;
    const float* Kg = kh + (size_t)bh * MRA_NB * MRA_HD;
    float lmax = -3e38f;
    for (int cc = 0; cc < 64; ++cc) {
        float acc = 0.f;
        const float* kr = Kg + (size_t)(coff + cc) * MRA_HD;
#pragma unroll 8
        for (int d = 0; d < MRA_HD; ++d) acc += Qh[row * MRA_HD + d] * kr[d];
        acc *= 0.125f;
        dst[cc] = acc;
        lmax = fmaxf(lmax, acc);
    }
    rmx[t] = lmax;
    __syncthreads();
    float rm = fmaxf(rmx[row * 2], rmx[row * 2 + 1]);
    if ((t & 1) == 0) rowmax[(size_t)bh * MRA_NB + row] = rm;
    float tcq = tc[(size_t)bh * MRA_NB + row];
    for (int cc = 0; cc < 64; ++cc) {
        float tck = tc[(size_t)bh * MRA_NB + coff + cc];
        if (tcq * tck < 0.5f) dst[cc] -= 1e4f;
    }
}

// ---------------------------------------------------------------------------
// K4: top-512 selection per bh via 10-bit radix histogram on float keys.
// value = lrl - row_max + diagonal prior (DIAG_N=3 -> |qi-ki|<=1 gets +5000).
// ---------------------------------------------------------------------------
__device__ __forceinline__ float topk_val(const float* lrl, const float* rowmax,
                                          int bh, int i) {
    int qi = i >> 7, ki = i & (MRA_NB - 1);
    float v = lrl[((size_t)bh * MRA_NB + qi) * MRA_NB + ki]
            - rowmax[(size_t)bh * MRA_NB + qi];
    int dd = qi - ki; if (dd < 0) dd = -dd;
    if (dd <= 1) v += 5000.f;
    return v;
}

__global__ __launch_bounds__(256) void mra_topk(
    const float* __restrict__ lrl, const float* __restrict__ rowmax,
    int* __restrict__ idxArr, float* __restrict__ thresh)
{
    __shared__ unsigned int hist[1024];
    __shared__ unsigned int sBstar, sCnt;
    int bh = blockIdx.x, t = threadIdx.x;
    for (int i = t; i < 1024; i += 256) hist[i] = 0u;
    if (t == 0) sCnt = 0u;
    __syncthreads();
    for (int i = t; i < MRA_NB * MRA_NB; i += 256)
        atomicAdd(&hist[fkey(topk_val(lrl, rowmax, bh, i)) >> 22], 1u);
    __syncthreads();
    if (t == 0) {
        unsigned int cum = 0, bs = 0;
        for (int bno = 1023; bno >= 0; --bno) {
            unsigned int cc = hist[bno];
            if (cum + cc >= MRA_TOPK) { bs = (unsigned)bno; break; }
            cum += cc;
        }
        sBstar = bs;
    }
    __syncthreads();
    unsigned int bs = sBstar;
    for (int i = t; i < MRA_NB * MRA_NB; i += 256) {       // strictly above cut bin
        float v = topk_val(lrl, rowmax, bh, i);
        if ((fkey(v) >> 22) > bs) {
            unsigned int p = atomicAdd(&sCnt, 1u);
            idxArr[(size_t)bh * MRA_TOPK + p] = i;
            atomicMinFloat(&thresh[bh], v);
        }
    }
    __syncthreads();
    for (int i = t; i < MRA_NB * MRA_NB; i += 256) {       // fill from cut bin, capped
        float v = topk_val(lrl, rowmax, bh, i);
        if ((fkey(v) >> 22) == bs) {
            unsigned int p = atomicAdd(&sCnt, 1u);
            if (p < MRA_TOPK) {
                idxArr[(size_t)bh * MRA_TOPK + p] = i;
                atomicMinFloat(&thresh[bh], v);
            }
        }
    }
}

// ---------------------------------------------------------------------------
// K5/K6: per selected block. phase 0: 32x32 logits via WMMA, atomic row max.
// phase 1: logits -> exp -> LDS transpose -> attn^T @ V via WMMA -> atomic
// scatter into hr_out / hr_norm.  One wave (32 threads) per block.
// ---------------------------------------------------------------------------
__global__ __launch_bounds__(32) void mra_hr_block(
    const unsigned short* __restrict__ qb, const unsigned short* __restrict__ kb,
    const unsigned short* __restrict__ vb, const int* __restrict__ idxArr,
    const float* __restrict__ am, float* __restrict__ max_vals,
    float* __restrict__ hr_out, float* __restrict__ hr_norm, int phase)
{
    __shared__ unsigned short Qs[32][72];
    __shared__ unsigned short Ks[32][72];
    __shared__ unsigned short Vs[32][72];
    __shared__ unsigned short At[32][40];   // attn^T-major: [q][k] bf16
    __shared__ float normL[32];

    int n  = blockIdx.x & (MRA_TOPK - 1);
    int bh = blockIdx.x >> 9;
    int pair = idxArr[(size_t)bh * MRA_TOPK + n];
    int qi = pair >> 7, ki = pair & (MRA_NB - 1);
    int t = threadIdx.x, lo = t & 15, hi = t >> 4;

    {
        const uint4* qs = (const uint4*)(qb + ((size_t)bh * MRA_S + qi * 32 + t) * MRA_HD);
        const uint4* ks = (const uint4*)(kb + ((size_t)bh * MRA_S + ki * 32 + t) * MRA_HD);
        uint4* qd = (uint4*)&Qs[t][0];
        uint4* kd = (uint4*)&Ks[t][0];
#pragma unroll
        for (int j = 0; j < 8; ++j) { qd[j] = qs[j]; kd[j] = ks[j]; }
    }
    normL[t] = 0.f;
    __syncthreads();

    // logits: C[k][q] = kb @ qb^T, K-dim = 64 (2 steps of 32)
    v8f c[2][2] = {};
#pragma unroll
    for (int ks = 0; ks < 2; ++ks) {
        v16bf a[2], b[2];
#pragma unroll
        for (int mi = 0; mi < 2; ++mi) {
            int row = mi * 16 + lo;
#pragma unroll
            for (int v = 0; v < 8; ++v) {
                int kk = ks * 32 + 2 * v + ((v >= 4) ? 8 : 0) + hi * 8;
                a[mi][2 * v]     = us2bf(Ks[row][kk]);
                a[mi][2 * v + 1] = us2bf(Ks[row][kk + 1]);
            }
        }
#pragma unroll
        for (int ni = 0; ni < 2; ++ni) {
            int col = ni * 16 + lo;
#pragma unroll
            for (int v = 0; v < 8; ++v) {
                int kk = ks * 32 + 2 * v + ((v >= 4) ? 8 : 0) + hi * 8;
                b[ni][2 * v]     = us2bf(Qs[col][kk]);
                b[ni][2 * v + 1] = us2bf(Qs[col][kk + 1]);
            }
        }
#pragma unroll
        for (int mi = 0; mi < 2; ++mi)
#pragma unroll
            for (int ni = 0; ni < 2; ++ni)
                c[mi][ni] = __builtin_amdgcn_wmma_f32_16x16x32_bf16(
                    false, a[mi], false, b[ni], (short)0, c[mi][ni], false, false);
    }

    if (phase == 0) {
        // sparse row max over all selected blocks of this query row
#pragma unroll
        for (int ni = 0; ni < 2; ++ni) {
            float qm = -1e30f;
#pragma unroll
            for (int mi = 0; mi < 2; ++mi)
#pragma unroll
                for (int r = 0; r < 8; ++r) qm = fmaxf(qm, c[mi][ni][r] * 0.125f);
            qm = fmaxf(qm, __shfl_xor(qm, 16, 32));
            if (t < 16)
                atomicMaxFloat(&max_vals[(size_t)bh * MRA_S + qi * 32 + ni * 16 + lo], qm);
        }
        return;
    }

    int bb = bh / MRA_NH;
    float mvq[2];
#pragma unroll
    for (int ni = 0; ni < 2; ++ni)
        mvq[ni] = max_vals[(size_t)bh * MRA_S + qi * 32 + ni * 16 + lo];

#pragma unroll
    for (int mi = 0; mi < 2; ++mi)
#pragma unroll
        for (int ni = 0; ni < 2; ++ni) {
            float part = 0.f;
#pragma unroll
            for (int r = 0; r < 8; ++r) {
                int kl = mi * 16 + r + hi * 8;
                float mk = 1.f + am[(size_t)bb * MRA_S + ki * 32 + kl] * 1e-4f;
                float val = c[mi][ni][r] * 0.125f - mvq[ni] - 1e4f * (1.f - mk);
                float e = __expf(val);
                part += e;
                At[ni * 16 + lo][kl] = f2bf(e);
            }
            atomicAdd(&normL[ni * 16 + lo], part);
        }
    __syncthreads();
    {
        const uint4* vs = (const uint4*)(vb + ((size_t)bh * MRA_S + ki * 32 + t) * MRA_HD);
        uint4* vd = (uint4*)&Vs[t][0];
#pragma unroll
        for (int j = 0; j < 8; ++j) vd[j] = vs[j];
    }
    __syncthreads();

    // out[q][d] = attn^T @ V, K-dim = 32 (one wmma per tile)
#pragma unroll
    for (int mi2 = 0; mi2 < 2; ++mi2) {
        v16bf a2;
        int qrow = mi2 * 16 + lo;
#pragma unroll
        for (int v = 0; v < 8; ++v) {
            int kk = 2 * v + ((v >= 4) ? 8 : 0) + hi * 8;
            a2[2 * v]     = us2bf(At[qrow][kk]);
            a2[2 * v + 1] = us2bf(At[qrow][kk + 1]);
        }
#pragma unroll
        for (int ni2 = 0; ni2 < 4; ++ni2) {
            v16bf b2;
            int dcol = ni2 * 16 + lo;
#pragma unroll
            for (int v = 0; v < 8; ++v) {
                int kk = 2 * v + ((v >= 4) ? 8 : 0) + hi * 8;
                b2[2 * v]     = us2bf(Vs[kk][dcol]);
                b2[2 * v + 1] = us2bf(Vs[kk + 1][dcol]);
            }
            v8f c2 = {};
            c2 = __builtin_amdgcn_wmma_f32_16x16x32_bf16(
                false, a2, false, b2, (short)0, c2, false, false);
#pragma unroll
            for (int r = 0; r < 8; ++r) {
                int q = mi2 * 16 + r + hi * 8;
                int d = ni2 * 16 + lo;
                atomicAdd(&hr_out[((size_t)bh * MRA_S + qi * 32 + q) * MRA_HD + d], c2[r]);
            }
        }
    }
    atomicAdd(&hr_norm[(size_t)bh * MRA_S + qi * 32 + t], normL[t]);
}

// ---------------------------------------------------------------------------
// K7: low-res branch + hr/lr correction combine -> d_out (B,S,768).
// One block per (bh, query block), 256 threads.
// ---------------------------------------------------------------------------
__global__ __launch_bounds__(256) void mra_final(
    const float* __restrict__ lrl, const float* __restrict__ rowmax,
    const float* __restrict__ tc, const float* __restrict__ thresh,
    const float* __restrict__ vh, const float* __restrict__ hr_out,
    const float* __restrict__ hr_norm, const float* __restrict__ max_vals,
    const float* __restrict__ am, float* __restrict__ out)
{
    __shared__ float lrA[MRA_NB];
    __shared__ float lrO[MRA_HD];
    __shared__ float sNorm;
    int bh = blockIdx.x >> 7;
    int qb = blockIdx.x & (MRA_NB - 1);
    int t = threadIdx.x;
    float rm = rowmax[(size_t)bh * MRA_NB + qb];
    float th = thresh[bh];
    if (t < MRA_NB) {
        float a = lrl[((size_t)bh * MRA_NB + qb) * MRA_NB + t] - rm;
        int dd = qb - t; if (dd < 0) dd = -dd;
        float prior = (dd <= 1) ? 5000.f : 0.f;
        float hmask = (a + prior >= th) ? 1.f : 0.f;
        lrA[t] = __expf(a - 1e4f * hmask) * tc[(size_t)bh * MRA_NB + t];
    }
    __syncthreads();
    if (t == 0) {
        float s = 0.f;
        for (int m = 0; m < MRA_NB; ++m) s += lrA[m];
        sNorm = s;
    }
    if (t < MRA_HD) {
        float s = 0.f;
        for (int m = 0; m < MRA_NB; ++m)
            s += lrA[m] * vh[((size_t)bh * MRA_NB + m) * MRA_HD + t];
        lrO[t] = s;
    }
    __syncthreads();
    int b = bh / MRA_NH, h = bh % MRA_NH;
    for (int e = t; e < 32 * MRA_HD; e += 256) {
        int sl = e >> 6, d = e & 63;
        int s = qb * 32 + sl;
        float mval = 1.f + am[(size_t)b * MRA_S + s] * 1e-4f;
        float mv = max_vals[(size_t)bh * MRA_S + s];
        float lc = (rm - mv) * mval;
        float lrc = __expf(fminf(lc, 0.f));
        float hrc = __expf(-fmaxf(lc, 0.f));
        float num = hr_out[((size_t)bh * MRA_S + s) * MRA_HD + d] * hrc + lrO[d] * lrc;
        float den = hr_norm[(size_t)bh * MRA_S + s] * hrc + sNorm * lrc + 1e-6f;
        out[((size_t)b * MRA_S + s) * MRA_H + h * MRA_HD + d] = num / den * mval;
    }
}

// ---------------------------------------------------------------------------
// Host-side launcher.
// ---------------------------------------------------------------------------
extern "C" void kernel_launch(void* const* d_in, const int* in_sizes, int n_in,
                              void* d_out, int out_size, void* d_ws, size_t ws_size,
                              hipStream_t stream) {
    (void)in_sizes; (void)n_in; (void)out_size; (void)ws_size;
    const float* X  = (const float*)d_in[0];
    const float* am = (const float*)d_in[1];
    const float* Wq = (const float*)d_in[2];
    const float* bq = (const float*)d_in[3];
    const float* Wk = (const float*)d_in[4];
    const float* bk = (const float*)d_in[5];
    const float* Wv = (const float*)d_in[6];
    const float* bv = (const float*)d_in[7];
    float* out = (float*)d_out;

    char* ws = (char*)d_ws;
    size_t off = 0;
    auto alloc = [&](size_t bytes) -> char* {
        off = (off + 255) & ~(size_t)255;
        char* p = ws + off;
        off += bytes;
        return p;
    };
    const size_t qkvBytes = (size_t)MRA_BH * MRA_S * MRA_HD * sizeof(unsigned short);
    unsigned short* q_bf = (unsigned short*)alloc(qkvBytes);
    unsigned short* k_bf = (unsigned short*)alloc(qkvBytes);
    unsigned short* v_bf = (unsigned short*)alloc(qkvBytes);
    float* hr_out  = (float*)alloc((size_t)MRA_BH * MRA_S * MRA_HD * sizeof(float));
    float* lrl     = (float*)alloc((size_t)MRA_BH * MRA_NB * MRA_NB * sizeof(float));
    float* q_hat   = (float*)alloc((size_t)MRA_BH * MRA_NB * MRA_HD * sizeof(float));
    float* k_hat   = (float*)alloc((size_t)MRA_BH * MRA_NB * MRA_HD * sizeof(float));
    float* v_hat   = (float*)alloc((size_t)MRA_BH * MRA_NB * MRA_HD * sizeof(float));
    float* tc      = (float*)alloc((size_t)MRA_BH * MRA_NB * sizeof(float));
    float* rowmax  = (float*)alloc((size_t)MRA_BH * MRA_NB * sizeof(float));
    float* thresh  = (float*)alloc((size_t)MRA_BH * sizeof(float));
    int*   idxArr  = (int*)alloc((size_t)MRA_BH * MRA_TOPK * sizeof(int));
    float* max_vals = (float*)alloc((size_t)MRA_BH * MRA_S * sizeof(float));
    float* hr_norm  = (float*)alloc((size_t)MRA_BH * MRA_S * sizeof(float));

    mra_init<<<1024, 256, 0, stream>>>(hr_out, hr_norm, max_vals, thresh);

    dim3 gemmGrid(3 * MRA_H / 64, (MRA_B * MRA_S) / 128);
    mra_qkv_gemm<<<gemmGrid, 256, 0, stream>>>(X, am, Wq, bq, Wk, bk, Wv, bv,
                                               q_bf, k_bf, v_bf);

    mra_pool<<<MRA_BH * MRA_NB, 64, 0, stream>>>(q_bf, k_bf, v_bf, am,
                                                 tc, q_hat, k_hat, v_hat);

    mra_lowres<<<MRA_BH, 256, 0, stream>>>(q_hat, k_hat, tc, lrl, rowmax);

    mra_topk<<<MRA_BH, 256, 0, stream>>>(lrl, rowmax, idxArr, thresh);

    mra_hr_block<<<MRA_BH * MRA_TOPK, 32, 0, stream>>>(
        q_bf, k_bf, v_bf, idxArr, am, max_vals, hr_out, hr_norm, /*phase=*/0);
    mra_hr_block<<<MRA_BH * MRA_TOPK, 32, 0, stream>>>(
        q_bf, k_bf, v_bf, idxArr, am, max_vals, hr_out, hr_norm, /*phase=*/1);

    mra_final<<<MRA_BH * MRA_NB, 256, 0, stream>>>(
        lrl, rowmax, tc, thresh, v_hat, hr_out, hr_norm, max_vals, am, out);
}
